// DenseBlock_67095979098952
// MI455X (gfx1250) — matrix operand order repaired and checked by reading, not compiled, not run on hardware
//
#include <hip/hip_runtime.h>
#include <hip/hip_bf16.h>

// ---------------------------------------------------------------------------
// DenseNet block on MI455X (gfx1250, wave32, WMMA f32_16x16x32_f16)
//
//   feats lives in d_out (32, 640, 28, 28) f32; x is copied into ch 0..255,
//   each layer appends 32 channels. BN moments persist in workspace (they
//   never change for already-written channels); gamma/beta folded per layer
//   into scale/shift. Both convs run as implicit GEMMs through
//   v_wmma_f32_16x16x32_f16 with LDS-staged f16 operands (128-bit ds stores)
//   and f32 accumulators.
// ---------------------------------------------------------------------------

typedef __attribute__((ext_vector_type(16))) _Float16 v16h;
typedef __attribute__((ext_vector_type(8)))  _Float16 v8h;
typedef __attribute__((ext_vector_type(8)))  float    v8f;

#define NPIX    25088      // 32 * 28 * 28
#define HW      784        // 28 * 28
#define CTOT    640        // final channel count
#define BOT     128        // bottleneck channels
#define GROWTH  32

// ---------------------------------------------------------------------------
// Copy x (32,256,784) into the concat buffer (32,640,784), channels 0..255.
// ---------------------------------------------------------------------------
__global__ __launch_bounds__(256) void copy_x_kernel(const float* __restrict__ x,
                                                     float* __restrict__ out) {
  int i = blockIdx.x * 256 + threadIdx.x;
  if (i >= 32 * 256 * HW) return;
  int n   = i / (256 * HW);
  int rem = i - n * 256 * HW;
  out[n * CTOT * HW + rem] = x[i];
}

// ---------------------------------------------------------------------------
// BN moments (biased var over N,H,W) for channels [c0, c0+gridDim.x) of the
// strided concat buffer. mean/rstd written once; reused by every later layer.
// ---------------------------------------------------------------------------
__global__ __launch_bounds__(256) void bn_stats_feats_kernel(
    const float* __restrict__ feats, float* __restrict__ mean,
    float* __restrict__ rstd, int c0) {
  int c = c0 + blockIdx.x;
  int t = threadIdx.x;
  float s = 0.f, sq = 0.f;
  for (int e = t; e < NPIX; e += 256) {
    int n = e / HW;
    int hw = e - n * HW;
    float v = feats[(n * CTOT + c) * HW + hw];
    s += v; sq += v * v;
  }
  __shared__ float ss[256], sqs[256];
  ss[t] = s; sqs[t] = sq;
  __syncthreads();
  for (int off = 128; off > 0; off >>= 1) {
    if (t < off) { ss[t] += ss[t + off]; sqs[t] += sqs[t + off]; }
    __syncthreads();
  }
  if (t == 0) {
    float m   = ss[0] * (1.f / NPIX);
    float var = sqs[0] * (1.f / NPIX) - m * m;
    mean[c] = m;
    rstd[c] = rsqrtf(var + 1e-5f);
  }
}

// Same, but for the contiguous bottleneck buffer h[128][25088].
__global__ __launch_bounds__(256) void bn_stats_h_kernel(
    const float* __restrict__ h, float* __restrict__ mean,
    float* __restrict__ rstd) {
  int c = blockIdx.x;
  int t = threadIdx.x;
  const float* src = h + (size_t)c * NPIX;
  float s = 0.f, sq = 0.f;
  for (int e = t; e < NPIX; e += 256) {
    float v = src[e];
    s += v; sq += v * v;
  }
  __shared__ float ss[256], sqs[256];
  ss[t] = s; sqs[t] = sq;
  __syncthreads();
  for (int off = 128; off > 0; off >>= 1) {
    if (t < off) { ss[t] += ss[t + off]; sqs[t] += sqs[t + off]; }
    __syncthreads();
  }
  if (t == 0) {
    float m   = ss[0] * (1.f / NPIX);
    float var = sqs[0] * (1.f / NPIX) - m * m;
    mean[c] = m;
    rstd[c] = rsqrtf(var + 1e-5f);
  }
}

// ---------------------------------------------------------------------------
// Fold per-layer gamma/beta with persistent moments: scale = g*rstd,
// shift = b - mean*scale. One FMA on the GEMM hot path instead of 4 loads.
// ---------------------------------------------------------------------------
__global__ __launch_bounds__(256) void fold_params_kernel(
    const float* __restrict__ g, const float* __restrict__ bta,
    const float* __restrict__ mean, const float* __restrict__ rstd,
    float* __restrict__ scale, float* __restrict__ shift, int nch) {
  int c = blockIdx.x * 256 + threadIdx.x;
  if (c < nch) {
    float s = g[c] * rstd[c];
    scale[c] = s;
    shift[c] = bta[c] - mean[c] * s;
  }
}

// ---------------------------------------------------------------------------
// 1x1 bottleneck: h[128, NPIX] = W1[128, c_in] x relu(bn(feats))[c_in, NPIX]
// Block tile 64(M) x 128(N), K-step 32. 8 waves: (mi 0..3, wc 0..1); each
// wave: 1 A-fragment, 4 B-fragments, 4 WMMAs per K-step.
// Staging is 8-halfs-per-thread packed -> ds_store_b128.
// ---------------------------------------------------------------------------
__global__ __launch_bounds__(256) void gemm1x1_bnrelu_kernel(
    const float* __restrict__ feats, const float* __restrict__ w1,
    const float* __restrict__ scale, const float* __restrict__ shift,
    float* __restrict__ h, int c_in) {
  constexpr int BM = 64, BN = 128, KS = 32, WP = 40;  // 80B rows: 16B aligned
  __shared__ __attribute__((aligned(16))) _Float16 Ws[BM][WP];  // [m][k]
  __shared__ __attribute__((aligned(16))) _Float16 Xs[BN][WP];  // [n][k]

  const int t    = threadIdx.x;
  const int lane = t & 31;
  const int wave = t >> 5;
  const int mBlk = blockIdx.y;            // 0..1 (128 / BM)
  const int pix0 = blockIdx.x * BN;
  const int mi   = wave & 3;              // M sub-tile (0..3)
  const int wc   = wave >> 2;             // N wave-column (0..1)

  v8f acc[4] = {};

  for (int k0 = 0; k0 < c_in; k0 += KS) {
    __syncthreads();
    // --- stage W tile: each thread one (m, k-octet): 2x global_load_b128,
    //     cvt, one ds_store_b128 ---
    {
      int m  = t & 63;
      int ko = t >> 6;                    // 0..3
      const float* src = &w1[(mBlk * BM + m) * c_in + k0 + ko * 8];
      v8h pk;
      #pragma unroll
      for (int j = 0; j < 8; ++j) pk[j] = (_Float16)src[j];
      *(v8h*)&Ws[m][ko * 8] = pk;
      // prefetch next K-tile of weights (global_prefetch_b8)
      if (k0 + KS < c_in) __builtin_prefetch(src + KS, 0, 3);
    }
    // --- stage activation tile, fused BN+ReLU, packed b128 stores ---
    #pragma unroll
    for (int r = 0; r < 2; ++r) {
      int it = t + 256 * r;
      int n  = it & (BN - 1);
      int ko = it >> 7;                   // 0..3
      int pix = pix0 + n;
      int nn  = pix / HW;
      int hw  = pix - nn * HW;
      int cb  = k0 + ko * 8;
      const float* src = &feats[(nn * CTOT + cb) * HW + hw];
      v8h pk;
      #pragma unroll
      for (int j = 0; j < 8; ++j) {
        float v = fmaf(src[j * HW], scale[cb + j], shift[cb + j]);
        pk[j] = (_Float16)fmaxf(v, 0.f);
      }
      *(v8h*)&Xs[n][ko * 8] = pk;
    }
    __syncthreads();

    // --- A fragment (16x32 f16), ISA VGPR layout ---
    v16h a;
    {
      int row = mi * 16 + (lane & 15);
      int kg  = (lane >> 4) * 8;
      #pragma unroll
      for (int j = 0; j < 8; ++j) a[j]     = Ws[row][kg + j];
      #pragma unroll
      for (int j = 0; j < 8; ++j) a[8 + j] = Ws[row][kg + 16 + j];
    }
    // --- 4 B fragments + WMMAs ---
    #pragma unroll
    for (int tn = 0; tn < 4; ++tn) {
      int nt  = wc * 4 + tn;
      int col = nt * 16 + (lane & 15);
      int kb  = (lane >> 4) * 8;
      v16h b;
      #pragma unroll
      for (int j = 0; j < 16; ++j) b[j] = Xs[col][kb + (j & 7) + (j >> 3) * 16];
      acc[tn] = __builtin_amdgcn_wmma_f32_16x16x32_f16(
          false, a, false, b, (short)0, acc[tn], false, false);
    }
  }

  // --- store C: D layout M = r + (lane>>4)*8, N = lane&15 ---
  #pragma unroll
  for (int tn = 0; tn < 4; ++tn) {
    int nt = wc * 4 + tn;
    int mg = mBlk * BM + mi * 16 + (lane >> 4) * 8;
    int pg = pix0 + nt * 16 + (lane & 15);
    #pragma unroll
    for (int r = 0; r < 8; ++r)
      h[(size_t)(mg + r) * NPIX + pg] = acc[tn][r];
  }
}

// ---------------------------------------------------------------------------
// 3x3 conv (pad 1): out[c_base+g] = sum_{tap,c} W2[g,c,tap] *
//   relu(bn(h))[c, shifted pixel] — zero pad applied AFTER bn+relu (matches
//   reference). Implicit GEMM: K = 9 taps x 128. Block tile 32(M) x 128(N).
// ---------------------------------------------------------------------------
__global__ __launch_bounds__(256) void gemm3x3_bnrelu_kernel(
    const float* __restrict__ h, const float* __restrict__ w2,
    const float* __restrict__ scale, const float* __restrict__ shift,
    float* __restrict__ out, int c_base) {
  constexpr int BM = 32, BN = 128, KS = 32, WP = 40;
  __shared__ __attribute__((aligned(16))) _Float16 Ws[BM][WP];
  __shared__ __attribute__((aligned(16))) _Float16 Xs[BN][WP];

  const int t    = threadIdx.x;
  const int lane = t & 31;
  const int wave = t >> 5;
  const int pix0 = blockIdx.x * BN;
  const int mi   = wave & 1;              // 2 M sub-tiles
  const int wc   = wave >> 1;             // 4 N wave-columns, 2 tiles each

  v8f acc[2] = {};

  for (int tap = 0; tap < 9; ++tap) {
    const int dy = tap / 3 - 1;
    const int dx = tap - (tap / 3) * 3 - 1;
    for (int c0 = 0; c0 < BOT; c0 += KS) {
      __syncthreads();
      // --- stage W tile: 32x32 halfs = 128 octets, threads 0..127 ---
      if (t < 128) {
        int m  = t & 31;
        int ko = t >> 5;                  // 0..3
        v8h pk;
        #pragma unroll
        for (int j = 0; j < 8; ++j)
          pk[j] = (_Float16)w2[((m * BOT) + c0 + ko * 8 + j) * 9 + tap];
        *(v8h*)&Ws[m][ko * 8] = pk;
      }
      // --- stage shifted activations, fused BN+ReLU + zero pad ---
      #pragma unroll
      for (int r = 0; r < 2; ++r) {
        int it = t + 256 * r;
        int n  = it & (BN - 1);
        int ko = it >> 7;                 // 0..3
        int pix = pix0 + n;
        int nn = pix / HW;
        int hw = pix - nn * HW;
        int y = hw / 28;
        int x = hw - y * 28;
        int ys = y + dy, xs = x + dx;
        bool inb = ((unsigned)ys < 28u) && ((unsigned)xs < 28u);
        int cb = c0 + ko * 8;
        const float* src = h + (size_t)cb * NPIX + nn * HW + ys * 28 + xs;
        v8h pk;
        #pragma unroll
        for (int j = 0; j < 8; ++j) {
          float v = 0.f;
          if (inb) {
            v = fmaf(src[(size_t)j * NPIX], scale[cb + j], shift[cb + j]);
            v = fmaxf(v, 0.f);
          }
          pk[j] = (_Float16)v;
        }
        *(v8h*)&Xs[n][ko * 8] = pk;
      }
      __syncthreads();

      v16h a;
      {
        int row = mi * 16 + (lane & 15);
        int kg  = (lane >> 4) * 8;
        #pragma unroll
        for (int j = 0; j < 8; ++j) a[j]     = Ws[row][kg + j];
        #pragma unroll
        for (int j = 0; j < 8; ++j) a[8 + j] = Ws[row][kg + 16 + j];
      }
      #pragma unroll
      for (int tn = 0; tn < 2; ++tn) {
        int nt  = wc * 2 + tn;
        int col = nt * 16 + (lane & 15);
        int kb  = (lane >> 4) * 8;
        v16h b;
        #pragma unroll
        for (int j = 0; j < 16; ++j) b[j] = Xs[col][kb + (j & 7) + (j >> 3) * 16];
        acc[tn] = __builtin_amdgcn_wmma_f32_16x16x32_f16(
            false, a, false, b, (short)0, acc[tn], false, false);
      }
    }
  }

  // --- store into concat buffer channels [c_base, c_base+32) ---
  #pragma unroll
  for (int tn = 0; tn < 2; ++tn) {
    int nt = wc * 2 + tn;
    int m  = mi * 16 + (lane >> 4) * 8;
    int pg = pix0 + nt * 16 + (lane & 15);
    int nn = pg / HW;
    int hw = pg - nn * HW;
    #pragma unroll
    for (int r = 0; r < 8; ++r)
      out[((nn * CTOT) + c_base + m + r) * HW + hw] = acc[tn][r];
  }
}

// ---------------------------------------------------------------------------
// Host orchestration (all on `stream`, graph-capture safe: no allocs/syncs).
// d_in: [0]=x, then per layer i: g1,b1,w1,g2,b2,w2 at 1+6i..
// workspace (floats): mean_f[640] | rstd_f[640] | mean_h[128] | rstd_h[128] |
//   scale_f[640] | shift_f[640] | scale_h[128] | shift_h[128] | h[128*25088]
// ---------------------------------------------------------------------------
extern "C" void kernel_launch(void* const* d_in, const int* in_sizes, int n_in,
                              void* d_out, int out_size, void* d_ws, size_t ws_size,
                              hipStream_t stream) {
  (void)in_sizes; (void)n_in; (void)out_size; (void)ws_size;
  const float* x   = (const float*)d_in[0];
  float*       out = (float*)d_out;
  float*       ws  = (float*)d_ws;

  float* meanf  = ws;
  float* rstdf  = ws + 640;
  float* meanh  = ws + 1280;
  float* rstdh  = ws + 1408;
  float* scalef = ws + 1536;
  float* shiftf = ws + 2176;
  float* scaleh = ws + 2816;
  float* shifth = ws + 2944;
  float* hbuf   = ws + 3072;   // 128 * 25088 floats (~12.8 MB)

  // 1) seed concat buffer with x
  copy_x_kernel<<<(32 * 256 * HW + 255) / 256, 256, 0, stream>>>(x, out);

  // 2) BN moments for the initial 256 channels (computed once; later layers
  //    only add moments for their 32 new channels)
  bn_stats_feats_kernel<<<256, 256, 0, stream>>>(out, meanf, rstdf, 0);

  for (int i = 0; i < 12; ++i) {
    const int c_in = 256 + i * GROWTH;
    const float* g1 = (const float*)d_in[1 + 6 * i + 0];
    const float* b1 = (const float*)d_in[1 + 6 * i + 1];
    const float* w1 = (const float*)d_in[1 + 6 * i + 2];
    const float* g2 = (const float*)d_in[1 + 6 * i + 3];
    const float* b2 = (const float*)d_in[1 + 6 * i + 4];
    const float* w2 = (const float*)d_in[1 + 6 * i + 5];

    // fold this layer's gamma/beta with persistent moments
    fold_params_kernel<<<(c_in + 255) / 256, 256, 0, stream>>>(
        g1, b1, meanf, rstdf, scalef, shiftf, c_in);

    // BN+ReLU fused 1x1 bottleneck GEMM -> hbuf
    dim3 grid1(NPIX / 128, 128 / 64);
    gemm1x1_bnrelu_kernel<<<grid1, 256, 0, stream>>>(
        out, w1, scalef, shiftf, hbuf, c_in);

    // moments of the fresh bottleneck activations + fold g2/b2
    bn_stats_h_kernel<<<BOT, 256, 0, stream>>>(hbuf, meanh, rstdh);
    fold_params_kernel<<<1, 256, 0, stream>>>(
        g2, b2, meanh, rstdh, scaleh, shifth, BOT);

    // BN+ReLU fused 3x3 implicit GEMM -> concat channels [c_in, c_in+32)
    gemm3x3_bnrelu_kernel<<<NPIX / 128, 256, 0, stream>>>(
        hbuf, w2, scaleh, shifth, out, c_in);

    // moments only for the 32 newly appended channels
    if (i < 11)
      bn_stats_feats_kernel<<<GROWTH, 256, 0, stream>>>(out, meanf, rstdf, c_in);
  }
}
